// Spk_Attention_18803366822322
// MI455X (gfx1250) — matrix operand order repaired and checked by reading, not compiled, not run on hardware
//
#include <hip/hip_runtime.h>
#include <hip/hip_bf16.h>
#include <math.h>

#define Hdim 512
#define Bdim 512
#define Ndim 256
#define ASTRIDE 520   // bf16 elements per staged A row (512 + 8 pad -> kills 16-way LDS bank conflicts)

typedef __bf16 v16bf __attribute__((ext_vector_type(16)));
typedef float  v8f   __attribute__((ext_vector_type(8)));

// Branchless tanh: correct saturation at +/-inf, one v_exp_f32, no divergence.
__device__ __forceinline__ float fast_tanh(float x) {
    return 1.f - 2.f / (__expf(2.f * x) + 1.f);
}

// ---------------- Prep kernel 1: W_s (f32, [k][n]) -> W_sT (bf16, [n][k]) ----------------
__global__ void ws_transpose_bf16(const float* __restrict__ Ws, __bf16* __restrict__ WsT) {
    int i = blockIdx.x * 256 + threadIdx.x;          // i indexes output [n][k]
    int n = i >> 9;
    int k = i & (Hdim - 1);
    WsT[i] = (__bf16)Ws[k * Hdim + n];
}

// ---------------- Prep kernel 2: W_resp = true_resp @ W_r  (f32 [B][H]) ----------------
__global__ __launch_bounds__(256) void resp_proj(const float* __restrict__ true_resp,
                                                 const float* __restrict__ W_r,
                                                 float* __restrict__ Wresp) {
    int b = blockIdx.x;
    __shared__ float tr[Hdim];
    for (int i = threadIdx.x; i < Hdim; i += 256) tr[i] = true_resp[(size_t)b * Hdim + i];
    __syncthreads();
    for (int o = threadIdx.x; o < Hdim; o += 256) {
        float acc = 0.f;
        for (int k = 0; k < Hdim; ++k) acc += tr[k] * W_r[(size_t)k * Hdim + o]; // coalesced over o
        Wresp[(size_t)b * Hdim + o] = acc;
    }
}

// ---------------- Main fused attention kernel: one block per batch element ----------------
__global__ __launch_bounds__(256) void spk_attention_main(
    const float* __restrict__ spk_emb,   // [B][N][H] f32
    const float* __restrict__ mask,      // [B][N] f32
    const float* __restrict__ true_resp, // [B][H] f32
    const float* __restrict__ wvec,      // [H] f32
    const float* __restrict__ Wp,        // [H][H] f32 (row = out)
    const float* __restrict__ Wx,        // [H][H] f32 (row = out)
    const __bf16* __restrict__ WsT,      // [H_out][H_k] bf16
    const float* __restrict__ Wresp,     // [B][H] f32
    float* __restrict__ out_resp,        // [B][H]
    float* __restrict__ out_alpha)       // [B][N]
{
    const int b    = blockIdx.x;
    const int tid  = threadIdx.x;
    const int wave = tid >> 5;
    const int lane = tid & 31;
    const int hl   = lane >> 4;   // half-wave selector
    const int m    = lane & 15;

    __shared__ __attribute__((aligned(16))) __bf16 Astage[2][16 * ASTRIDE];  // double buffered
    __shared__ __attribute__((aligned(16))) float  scores[Ndim];
    __shared__ __attribute__((aligned(16))) float  red[256];
    __shared__ __attribute__((aligned(16))) float  alpha_s[Ndim];
    __shared__ __attribute__((aligned(16))) float  resp_pre[Hdim];
    __shared__ __attribute__((aligned(16))) float  tr_s[Hdim];

    for (int i = tid; i < Ndim; i += 256) scores[i] = 0.f;
    for (int i = tid; i < Hdim; i += 256) tr_s[i] = true_resp[(size_t)b * Hdim + i];

    const float* semb_b = spk_emb + (size_t)b * Ndim * Hdim;

    // ---- Stage 1: scores[n] = w . tanh(spk_emb[b] @ W_s + W_resp[b])  via bf16 WMMA ----
    // Each 16xH strip of N rows is contiguous in memory (strip rt starts at semb_b + rt*16*H).
    // Thread tid owns chunks c = tid + 256*j (j=0..3), each chunk = 8 consecutive f32 ->
    // 2x global_load_b128 -> 8 cvt -> 1x ds_store_b128.

    // Prologue: stage strip 0 into buffer 0.
    {
        const float* src = semb_b;                    // strip 0
        #pragma unroll
        for (int j = 0; j < 4; ++j) {
            int c = tid + 256 * j;
            int r = c >> 6, k = (c & 63) << 3;
            float4 g0 = *(const float4*)(src + c * 8);
            float4 g1 = *(const float4*)(src + c * 8 + 4);
            union { __bf16 h[8]; uint4 q; } pk;
            pk.h[0] = (__bf16)g0.x; pk.h[1] = (__bf16)g0.y;
            pk.h[2] = (__bf16)g0.z; pk.h[3] = (__bf16)g0.w;
            pk.h[4] = (__bf16)g1.x; pk.h[5] = (__bf16)g1.y;
            pk.h[6] = (__bf16)g1.z; pk.h[7] = (__bf16)g1.w;
            *(uint4*)&Astage[0][r * ASTRIDE + k] = pk.q;
        }
    }
    __syncthreads();

    int cur = 0;
    for (int rt = 0; rt < 16; ++rt) {                 // 16 row tiles of N
        const int n0 = rt * 16;

        // Issue next strip's global loads early: latency hidden under the 64 WMMAs below.
        float4 g[4][2];
        if (rt < 15) {
            const float* src = semb_b + (size_t)(n0 + 16) * Hdim;
            #pragma unroll
            for (int j = 0; j < 4; ++j) {
                int c = tid + 256 * j;
                g[j][0] = *(const float4*)(src + c * 8);
                g[j][1] = *(const float4*)(src + c * 8 + 4);
            }
        }

        // ---- compute: 4 column tiles per wave, one shared A fragment, 4 indep acc chains
        const __bf16* arow = &Astage[cur][m * ASTRIDE];
        const int hbase = wave * 64 + m;              // hcol(cti) = hbase + cti*16
        v8f acc0 = {}, acc1 = {}, acc2 = {}, acc3 = {};
        const __bf16* bbase = WsT + (size_t)hbase * Hdim + hl * 16;

        #pragma unroll 2
        for (int kt = 0; kt < 16; ++kt) {
            const int k0 = kt * 32;
            // A fragment: contiguous K order k = 16*hl + s (pairs of K per VGPR; merges to 2x ds_load_b128)
            union { v16bf v; unsigned int u[8]; } af;
            #pragma unroll
            for (int p = 0; p < 8; ++p) {
                int kp = k0 + hl * 16 + 2 * p;
                af.u[p] = *(const unsigned int*)(&arow[kp]);
            }
            // 4 B fragments: lane = column, 16 consecutive K values (32B each)
            v16bf bf0 = *(const v16bf*)(bbase + 0 * 16 * Hdim + k0);
            v16bf bf1 = *(const v16bf*)(bbase + 1 * 16 * Hdim + k0);
            v16bf bf2 = *(const v16bf*)(bbase + 2 * 16 * Hdim + k0);
            v16bf bf3 = *(const v16bf*)(bbase + 3 * 16 * Hdim + k0);
            acc0 = __builtin_amdgcn_wmma_f32_16x16x32_bf16(false, af.v, false, bf0, (short)0, acc0, false, false);
            acc1 = __builtin_amdgcn_wmma_f32_16x16x32_bf16(false, af.v, false, bf1, (short)0, acc1, false, false);
            acc2 = __builtin_amdgcn_wmma_f32_16x16x32_bf16(false, af.v, false, bf2, (short)0, acc2, false, false);
            acc3 = __builtin_amdgcn_wmma_f32_16x16x32_bf16(false, af.v, false, bf3, (short)0, acc3, false, false);
        }

        // epilogue: per-lane column h is constant per cti -> scalar W_resp[h], w[h]
        float accs[8];
        #pragma unroll
        for (int r = 0; r < 8; ++r) accs[r] = 0.f;
        {
            const float* wrp = Wresp + (size_t)b * Hdim + hbase;
            const float wr0 = wrp[0], wr1 = wrp[16], wr2 = wrp[32], wr3 = wrp[48];
            const float wh0 = wvec[hbase], wh1 = wvec[hbase + 16];
            const float wh2 = wvec[hbase + 32], wh3 = wvec[hbase + 48];
            #pragma unroll
            for (int r = 0; r < 8; ++r) {             // row n = n0 + hl*8 + r
                accs[r] += fast_tanh(acc0[r] + wr0) * wh0;
                accs[r] += fast_tanh(acc1[r] + wr1) * wh1;
                accs[r] += fast_tanh(acc2[r] + wr2) * wh2;
                accs[r] += fast_tanh(acc3[r] + wr3) * wh3;
            }
        }
        // reduce the 16 columns held across the 16 lanes of each half-wave
        #pragma unroll
        for (int r = 0; r < 8; ++r) {
            float s = accs[r];
            for (int off = 1; off < 16; off <<= 1) s += __shfl_xor(s, off, 32);
            accs[r] = s;
        }
        if (m == 0) {                                  // lanes 0 and 16 commit rows
            #pragma unroll
            for (int r = 0; r < 8; ++r)
                atomicAdd(&scores[n0 + hl * 8 + r], accs[r]);   // ds_add_f32
        }

        // Convert + store the prefetched strip into the alternate buffer.
        if (rt < 15) {
            #pragma unroll
            for (int j = 0; j < 4; ++j) {
                int c = tid + 256 * j;
                int r = c >> 6, k = (c & 63) << 3;
                union { __bf16 h[8]; uint4 q; } pk;
                pk.h[0] = (__bf16)g[j][0].x; pk.h[1] = (__bf16)g[j][0].y;
                pk.h[2] = (__bf16)g[j][0].z; pk.h[3] = (__bf16)g[j][0].w;
                pk.h[4] = (__bf16)g[j][1].x; pk.h[5] = (__bf16)g[j][1].y;
                pk.h[6] = (__bf16)g[j][1].z; pk.h[7] = (__bf16)g[j][1].w;
                *(uint4*)&Astage[cur ^ 1][r * ASTRIDE + k] = pk.q;
            }
        }
        __syncthreads();
        cur ^= 1;
    }

    // ---- Stage 2: softmax over N (one thread per n) ----
    float sc = scores[tid] * mask[(size_t)b * Ndim + tid];
    red[tid] = sc;
    __syncthreads();
    for (int off = 128; off > 0; off >>= 1) {
        if (tid < off) red[tid] = fmaxf(red[tid], red[tid + off]);
        __syncthreads();
    }
    const float mx = red[0];
    __syncthreads();
    const float e = __expf(sc - mx);
    red[tid] = e;
    __syncthreads();
    for (int off = 128; off > 0; off >>= 1) {
        if (tid < off) red[tid] += red[tid + off];
        __syncthreads();
    }
    const float a = e / red[0];
    alpha_s[tid] = a;
    out_alpha[(size_t)b * Ndim + tid] = a;
    __syncthreads();

    // ---- Stage 3: resp_pre[h] = sum_n alpha[n] * spk_emb[b][n][h]  (float2 per thread) ----
    {
        float2 accv = make_float2(0.f, 0.f);
        const float* pb = semb_b + 2 * tid;
        for (int n = 0; n < Ndim; ++n) {
            const float an = alpha_s[n];
            const float2 v = *(const float2*)(pb + (size_t)n * Hdim);
            accv.x += an * v.x;
            accv.y += an * v.y;
        }
        resp_pre[2 * tid]     = accv.x;
        resp_pre[2 * tid + 1] = accv.y;
    }
    __syncthreads();

    // ---- Stage 4: out = tanh(resp_pre @ Wp^T + true_resp @ Wx^T) ----
    // One output per wave; lanes split the 512-dot into float4 chunks (fully coalesced).
    for (int o = wave; o < Hdim; o += 8) {
        const float* wp = Wp + (size_t)o * Hdim;
        const float* wx = Wx + (size_t)o * Hdim;
        float acc = 0.f;
        #pragma unroll
        for (int j = 0; j < 4; ++j) {
            int k = j * 128 + lane * 4;
            float4 aa = *(const float4*)&resp_pre[k];
            float4 cc = *(const float4*)(wp + k);
            float4 dd = *(const float4*)&tr_s[k];
            float4 ee = *(const float4*)(wx + k);
            acc += aa.x * cc.x + aa.y * cc.y + aa.z * cc.z + aa.w * cc.w;
            acc += dd.x * ee.x + dd.y * ee.y + dd.z * ee.z + dd.w * ee.w;
        }
        for (int off = 1; off < 32; off <<= 1) acc += __shfl_xor(acc, off, 32);
        if (lane == 0) out_resp[(size_t)b * Hdim + o] = fast_tanh(acc);
    }
}

extern "C" void kernel_launch(void* const* d_in, const int* in_sizes, int n_in,
                              void* d_out, int out_size, void* d_ws, size_t ws_size,
                              hipStream_t stream) {
    const float* spk_emb   = (const float*)d_in[0];
    const float* mask      = (const float*)d_in[1];
    const float* true_resp = (const float*)d_in[2];
    const float* W_s       = (const float*)d_in[3];
    const float* W_r       = (const float*)d_in[4];
    const float* wvec      = (const float*)d_in[5];
    const float* Wp        = (const float*)d_in[6];
    const float* Wx        = (const float*)d_in[7];

    __bf16* WsT   = (__bf16*)d_ws;                                   // H*H*2   = 512 KB
    float*  Wresp = (float*)((char*)d_ws + (size_t)Hdim * Hdim * 2); // B*H*4   = 1 MB

    float* out_resp  = (float*)d_out;                  // [B][H]
    float* out_alpha = out_resp + (size_t)Bdim * Hdim; // [B][N]

    ws_transpose_bf16<<<(Hdim * Hdim) / 256, 256, 0, stream>>>(W_s, WsT);
    resp_proj<<<Bdim, 256, 0, stream>>>(true_resp, W_r, Wresp);
    spk_attention_main<<<Bdim, 256, 0, stream>>>(spk_emb, mask, true_resp, wvec,
                                                 Wp, Wx, WsT, Wresp,
                                                 out_resp, out_alpha);
}
